// SAttention_49692771614988
// MI455X (gfx1250) — matrix operand order repaired and checked by reading, not compiled, not run on hardware
//
#include <hip/hip_runtime.h>

typedef __attribute__((ext_vector_type(16))) __bf16 v16bf;
typedef __attribute__((ext_vector_type(8)))  float  v8f;

#define B_  32
#define L_  2048
#define U_  512
#define H_  8
#define HD_ 64
#define KD  1536          // 3 * U
#define MT  128           // positions per workgroup tile
#define NT  32            // channels per workgroup tile
#define KSTEPS 48         // KD / 32

// ---- workspace layout (bytes) ----
constexpr size_t SZ_WT = (size_t)3 * KD * U_ * 2;          // bf16 fragment-packed weights
constexpr size_t SZ_KV = (size_t)B_ * L_ * U_ * 2;         // bf16 k or v
constexpr size_t SZ_MX = (size_t)B_ * U_ * 4;              // int-keyed channel max
constexpr size_t SZ_S  = (size_t)B_ * H_ * 2049 * 4;       // fp32 scores
constexpr size_t OFF_WT = 0;
constexpr size_t OFF_K  = OFF_WT + SZ_WT;
constexpr size_t OFF_V  = OFF_K + SZ_KV;
constexpr size_t OFF_QM = OFF_V + SZ_KV;
constexpr size_t OFF_KM = OFF_QM + SZ_MX;
constexpr size_t OFF_VM = OFF_KM + SZ_MX;
constexpr size_t OFF_S  = OFF_VM + SZ_MX;
constexpr size_t OFF_Z  = OFF_S + SZ_S;

__device__ __forceinline__ int fkey(float f) {          // monotone float -> int
  int b = __float_as_int(f);
  return b >= 0 ? b : (b ^ 0x7fffffff);
}
__device__ __forceinline__ float fval(int k) {          // inverse
  int b = k >= 0 ? k : (k ^ 0x7fffffff);
  return __int_as_float(b);
}
__device__ __forceinline__ float bflo(unsigned u) { return __uint_as_float(u << 16); }
__device__ __forceinline__ float bfhi(unsigned u) { return __uint_as_float(u & 0xffff0000u); }

#define RSQRT_U 0.044194173824159216f                    // 1/sqrt(512)

// ---------------------------------------------------------------------------
// Kernel 1: repack fp32 conv weights into bf16 *fragment-native* layout:
//   Wt[((c*48 + ks)*512 + n)*32 + hf*16 + e]  holds  W_c[n, i, tap]
// with K-index j = ks*32 + (hf*16 + e), tap = j/512, i = j%512.
// Each lane's B fragment is then one contiguous 32-byte chunk.
// Also inits the three channel-max buffers.
// ---------------------------------------------------------------------------
__global__ __launch_bounds__(256)
void prep_kernel(const float* __restrict__ Wq, const float* __restrict__ Wk,
                 const float* __restrict__ Wv, __bf16* __restrict__ Wt,
                 int* __restrict__ qm, int* __restrict__ km, int* __restrict__ vm) {
  int idx = blockIdx.x * 256 + threadIdx.x;
  if (idx < 3 * KSTEPS * U_ * 32) {
    int ee  = idx & 31;             // hf*16 + e
    int n   = (idx >> 5) & 511;     // output channel
    int cks = idx >> 14;            // c*48 + ks
    int c   = cks / KSTEPS;
    int ks  = cks % KSTEPS;
    int j   = ks * 32 + ee;         // im2col K index
    int tap = j / U_;
    int i   = j % U_;
    const float* W = (c == 0) ? Wq : (c == 1) ? Wk : Wv;
    Wt[idx] = (__bf16)W[(size_t)n * KD + i * 3 + tap];
  }
  if (idx < 3 * B_ * U_) {
    int which = idx / (B_ * U_);
    int r = idx % (B_ * U_);
    int* p = (which == 0) ? qm : (which == 1) ? km : vm;
    p[r] = (int)0x80000000;
  }
}

// ---------------------------------------------------------------------------
// Kernel 2: fused q/k/v conv-GEMM. Workgroup tile 128 pos x 32 chan; each
// wave owns 32 pos x 16 chan = two 16x16 M-subtiles, so the 3 B fragments
// (one 32-B global vector each) feed 6 WMMAs. A staged in LDS
// (double-buffered, b128-vectorized).
// ---------------------------------------------------------------------------
__global__ __launch_bounds__(256)
void qkv_gemm(const float* __restrict__ t, const __bf16* __restrict__ Wt,
              const float* __restrict__ bq, const float* __restrict__ bk,
              const float* __restrict__ bv,
              __bf16* __restrict__ Kbuf, __bf16* __restrict__ Vbuf,
              int* __restrict__ qm, int* __restrict__ km, int* __restrict__ vm) {
  const int b  = blockIdx.z;
  const int l0 = blockIdx.x * MT;
  const int n0 = blockIdx.y * NT;
  const int tid  = threadIdx.x;
  const int lane = tid & 31;
  const int wave = tid >> 5;        // 0..7
  const int hf   = lane >> 4;       // half-wave
  const int l16  = lane & 15;
  const int wm   = wave & 3;        // position sub-tile (32 rows each)
  const int wn   = wave >> 2;       // channel sub-tile (16 cols each)

  // row stride 40 elements (80 B): 16-B aligned chunks, clean b128 ds ops
  __shared__ __align__(16) __bf16 As[2][MT][40];

  v8f accq0 = {}, accq1 = {};
  v8f acck0 = {}, acck1 = {};
  v8f accv0 = {}, accv1 = {};

  const float* tb = t + (size_t)b * L_ * U_;

  // per-lane B fragment pointers (32-B chunks, advance 512*32 elems per K-step)
  const int ocol = n0 + wn * 16 + l16;                  // global channel
  const __bf16* wqp = Wt + ((size_t)(0 * KSTEPS) * U_ + ocol) * 32 + hf * 16;
  const __bf16* wkp = Wt + ((size_t)(1 * KSTEPS) * U_ + ocol) * 32 + hf * 16;
  const __bf16* wvp = Wt + ((size_t)(2 * KSTEPS) * U_ + ocol) * 32 + hf * 16;

  auto stage = [&](int ks, int bufi) {
    const int kk0 = ks * 32;
    const int tap = kk0 >> 9;       // constant within a 32-step
    const int i0  = kk0 & 511;
    const int m   = tid >> 1;            // 0..127
    const int kkb = (tid & 1) * 16;      // 0 or 16
    const int l   = l0 + m + tap - 1;
    float4 f0 = {0.f, 0.f, 0.f, 0.f}, f1 = f0, f2 = f0, f3 = f0;
    if (l >= 0 && l < L_) {
      const float* src = tb + (size_t)l * U_ + i0 + kkb;
      f0 = *(const float4*)src;
      f1 = *(const float4*)(src + 4);
      f2 = *(const float4*)(src + 8);
      f3 = *(const float4*)(src + 12);
    }
    __align__(16) __bf16 tmp[16];
    tmp[0]  = (__bf16)f0.x; tmp[1]  = (__bf16)f0.y;
    tmp[2]  = (__bf16)f0.z; tmp[3]  = (__bf16)f0.w;
    tmp[4]  = (__bf16)f1.x; tmp[5]  = (__bf16)f1.y;
    tmp[6]  = (__bf16)f1.z; tmp[7]  = (__bf16)f1.w;
    tmp[8]  = (__bf16)f2.x; tmp[9]  = (__bf16)f2.y;
    tmp[10] = (__bf16)f2.z; tmp[11] = (__bf16)f2.w;
    tmp[12] = (__bf16)f3.x; tmp[13] = (__bf16)f3.y;
    tmp[14] = (__bf16)f3.z; tmp[15] = (__bf16)f3.w;
    *(uint4*)&As[bufi][m][kkb]     = *(const uint4*)&tmp[0];
    *(uint4*)&As[bufi][m][kkb + 8] = *(const uint4*)&tmp[8];
  };

  stage(0, 0);
  __syncthreads();

  const int arow0 = wm * 32 + l16;
  const int arow1 = arow0 + 16;
  int buf = 0;
  for (int ks = 0; ks < KSTEPS; ++ks) {
    // two A fragments: one contiguous 32-B run each (k = 16*hf + s, matches B)
    union { uint4 q[2]; v16bf v; } ua0, ua1;
    ua0.q[0] = *(const uint4*)&As[buf][arow0][16 * hf];
    ua0.q[1] = *(const uint4*)&As[buf][arow0][16 * hf + 8];
    ua1.q[0] = *(const uint4*)&As[buf][arow1][16 * hf];
    ua1.q[1] = *(const uint4*)&As[buf][arow1][16 * hf + 8];
    const v16bf a0 = ua0.v;
    const v16bf a1 = ua1.v;
    // B fragments: one contiguous 32-B vector load each, shared by 2 M-tiles
    const v16bf fq = *(const v16bf*)(wqp + (size_t)ks * (U_ * 32));
    const v16bf fk = *(const v16bf*)(wkp + (size_t)ks * (U_ * 32));
    const v16bf fv = *(const v16bf*)(wvp + (size_t)ks * (U_ * 32));

    accq0 = __builtin_amdgcn_wmma_f32_16x16x32_bf16(false, a0, false, fq, (short)0, accq0, false, false);
    acck0 = __builtin_amdgcn_wmma_f32_16x16x32_bf16(false, a0, false, fk, (short)0, acck0, false, false);
    accv0 = __builtin_amdgcn_wmma_f32_16x16x32_bf16(false, a0, false, fv, (short)0, accv0, false, false);
    accq1 = __builtin_amdgcn_wmma_f32_16x16x32_bf16(false, a1, false, fq, (short)0, accq1, false, false);
    acck1 = __builtin_amdgcn_wmma_f32_16x16x32_bf16(false, a1, false, fk, (short)0, acck1, false, false);
    accv1 = __builtin_amdgcn_wmma_f32_16x16x32_bf16(false, a1, false, fv, (short)0, accv1, false, false);

    if (ks + 1 < KSTEPS) stage(ks + 1, buf ^ 1);
    __syncthreads();
    buf ^= 1;
  }

  // epilogue: bias, k/v store (bf16), q/k/v channel-max reduce + atomics
  const float biasq = bq[ocol], biask = bk[ocol], biasv = bv[ocol];
  float mq = -__builtin_inff(), mk = -__builtin_inff(), mv = -__builtin_inff();
#pragma unroll
  for (int st = 0; st < 2; ++st) {
    const v8f aq = st ? accq1 : accq0;
    const v8f ak = st ? acck1 : acck0;
    const v8f av = st ? accv1 : accv0;
#pragma unroll
    for (int r = 0; r < 8; ++r) {
      const int m = (hf ? (8 + r) : r) + 16 * st;        // C/D row mapping
      const int l = l0 + wm * 32 + m;
      const float qv = aq[r] + biasq;
      const float kv = ak[r] + biask;
      const float vv = av[r] + biasv;
      mq = fmaxf(mq, qv);
      mk = fmaxf(mk, kv);
      mv = fmaxf(mv, vv);
      const size_t off = ((size_t)b * L_ + l) * U_ + ocol;
      Kbuf[off] = (__bf16)kv;
      Vbuf[off] = (__bf16)vv;
    }
  }
  mq = fmaxf(mq, __shfl_xor(mq, 16, 32));
  mk = fmaxf(mk, __shfl_xor(mk, 16, 32));
  mv = fmaxf(mv, __shfl_xor(mv, 16, 32));
  if (hf == 0) {
    atomicMax(&qm[b * U_ + ocol], fkey(mq));
    atomicMax(&km[b * U_ + ocol], fkey(mk));
    atomicMax(&vm[b * U_ + ocol], fkey(mv));
  }
}

// ---------------------------------------------------------------------------
// Kernel 3: masked scaled scores vs on-the-fly maxpool3(k). b128 loads +
// shift-unpack (bits<<16 is exact bf16->f32). Grid (L/256, H, B).
// ---------------------------------------------------------------------------
__global__ __launch_bounds__(256)
void scores_kernel(const __bf16* __restrict__ Kbuf, const int* __restrict__ qm,
                   const int* __restrict__ t_mask, float* __restrict__ S) {
  const int b = blockIdx.z, h = blockIdx.y;
  const int l = blockIdx.x * 256 + threadIdx.x;
  __shared__ float qsh[HD_];
  if (threadIdx.x < HD_)
    qsh[threadIdx.x] = fval(qm[b * U_ + h * HD_ + threadIdx.x]);
  __syncthreads();

  float s;
  if (t_mask[b * L_ + l] == 0) {
    s = -__builtin_inff();
  } else {
    const uint4* r0 = (const uint4*)(Kbuf + ((size_t)b * L_ + l) * U_ + h * HD_);
    const int rs = U_ / 8;          // uint4 row stride
    const bool hasm = (l > 0), hasp = (l < L_ - 1);
    float acc = 0.0f;
#pragma unroll
    for (int c = 0; c < 8; ++c) {
      const uint4 a0 = r0[c];
      const uint4 am = hasm ? r0[c - rs] : a0;   // max(a,a)=a for missing edge
      const uint4 ap = hasp ? r0[c + rs] : a0;
      const unsigned u0[4] = {a0.x, a0.y, a0.z, a0.w};
      const unsigned um[4] = {am.x, am.y, am.z, am.w};
      const unsigned up[4] = {ap.x, ap.y, ap.z, ap.w};
#pragma unroll
      for (int w = 0; w < 4; ++w) {
        const float lo = fmaxf(fmaxf(bflo(u0[w]), bflo(um[w])), bflo(up[w]));
        const float hi = fmaxf(fmaxf(bfhi(u0[w]), bfhi(um[w])), bfhi(up[w]));
        acc += qsh[c * 8 + w * 2]     * lo;
        acc += qsh[c * 8 + w * 2 + 1] * hi;
      }
    }
    s = acc * RSQRT_U;
  }
  S[((size_t)b * H_ + h) * 2049 + 1 + l] = s;
}

// ---------------------------------------------------------------------------
// Kernel 4: per (b,h): global-key score, softmax over 2049, weighted
// maxpool3(v) + global value, g_mask. Grid (H, B), 256 threads.
// ---------------------------------------------------------------------------
__global__ __launch_bounds__(256)
void attn_kernel(const __bf16* __restrict__ Vbuf, const int* __restrict__ qm,
                 const int* __restrict__ km, const int* __restrict__ vm,
                 const int* __restrict__ g_mask, float* __restrict__ S,
                 float* __restrict__ Z) {
  const int h = blockIdx.x, b = blockIdx.y;
  const int tid = threadIdx.x;
  float* s = S + ((size_t)b * H_ + h) * 2049;

  __shared__ float sh[256];
  __shared__ float qsh[HD_], kg[HD_], vg[HD_];
  if (tid < HD_) {
    qsh[tid] = fval(qm[b * U_ + h * HD_ + tid]);
    kg[tid]  = fval(km[b * U_ + h * HD_ + tid]);
    vg[tid]  = fval(vm[b * U_ + h * HD_ + tid]);
  }
  __syncthreads();

  // score for the global key
  sh[tid] = (tid < HD_) ? qsh[tid] * kg[tid] : 0.0f;
  __syncthreads();
  for (int st = 128; st > 0; st >>= 1) { if (tid < st) sh[tid] += sh[tid + st]; __syncthreads(); }
  float s0 = sh[0] * RSQRT_U;
  if (g_mask[b] == 0) s0 = -__builtin_inff();
  if (tid == 0) s[0] = s0;
  __syncthreads();

  // softmax max
  float mx = -__builtin_inff();
  for (int i = tid; i < 2049; i += 256) mx = fmaxf(mx, s[i]);
  sh[tid] = mx; __syncthreads();
  for (int st = 128; st > 0; st >>= 1) { if (tid < st) sh[tid] = fmaxf(sh[tid], sh[tid + st]); __syncthreads(); }
  mx = sh[0]; __syncthreads();

  // softmax denom
  float sm = 0.0f;
  for (int i = tid; i < 2049; i += 256) sm += __expf(s[i] - mx);
  sh[tid] = sm; __syncthreads();
  for (int st = 128; st > 0; st >>= 1) { if (tid < st) sh[tid] += sh[tid + st]; __syncthreads(); }
  const float inv = 1.0f / sh[0];
  __syncthreads();

  // weighted maxpool3(v): 4 l-groups x 64 dims
  const int g = tid >> 6, d = tid & 63;
  const __bf16* vb = Vbuf + (size_t)b * L_ * U_ + h * HD_ + d;
  float acc = 0.0f;
  for (int l = g; l < L_; l += 4) {
    const float w = __expf(s[1 + l] - mx);
    if (w != 0.0f) {
      float vv = (float)vb[(size_t)l * U_];
      if (l > 0)      vv = fmaxf(vv, (float)vb[(size_t)(l - 1) * U_]);
      if (l < L_ - 1) vv = fmaxf(vv, (float)vb[(size_t)(l + 1) * U_]);
      acc += w * vv;
    }
  }
  sh[tid] = acc;
  __syncthreads();
  if (tid < HD_) {
    const float r  = sh[tid] + sh[tid + 64] + sh[tid + 128] + sh[tid + 192];
    const float w0 = __expf(s[0] - mx);
    const float gm = (float)g_mask[b];
    Z[(size_t)b * U_ + h * HD_ + tid] = (r + w0 * vg[tid]) * inv * gm;
  }
}

// ---------------------------------------------------------------------------
// Kernel 5: 8x8 head mix + 512x512 output projection. Grid (B), 256 threads.
// ---------------------------------------------------------------------------
__global__ __launch_bounds__(256)
void final_kernel(const float* __restrict__ Z, const float* __restrict__ Wh,
                  const float* __restrict__ bh, const float* __restrict__ Wo,
                  const float* __restrict__ bo, float* __restrict__ out) {
  const int b = blockIdx.x, tid = threadIdx.x;
  __shared__ float z[U_];
  __shared__ float y[U_];
  for (int i = tid; i < U_; i += 256) z[i] = Z[(size_t)b * U_ + i];
  __syncthreads();
  for (int i = tid; i < U_; i += 256) {
    const int hp = i >> 6, d = i & 63;
    float acc = bh[hp];
#pragma unroll
    for (int hh = 0; hh < H_; ++hh) acc += Wh[hp * H_ + hh] * z[hh * HD_ + d];
    y[i] = acc;
  }
  __syncthreads();
  for (int o = tid; o < U_; o += 256) {
    float acc = bo[o];
    const float* w = Wo + (size_t)o * U_;
    for (int i = 0; i < U_; ++i) acc += w[i] * y[i];
    out[(size_t)b * U_ + o] = acc;
  }
}

// ---------------------------------------------------------------------------
extern "C" void kernel_launch(void* const* d_in, const int* in_sizes, int n_in,
                              void* d_out, int out_size, void* d_ws, size_t ws_size,
                              hipStream_t stream) {
  const float* t      = (const float*)d_in[0];
  // d_in[1] (g) is unused by the reference computation
  const int*   g_mask = (const int*)d_in[2];
  const int*   t_mask = (const int*)d_in[3];
  const float* Wq = (const float*)d_in[4];
  const float* bq = (const float*)d_in[5];
  const float* Wk = (const float*)d_in[6];
  const float* bk = (const float*)d_in[7];
  const float* Wv = (const float*)d_in[8];
  const float* bv = (const float*)d_in[9];
  const float* Wh = (const float*)d_in[10];
  const float* bh = (const float*)d_in[11];
  const float* Wo = (const float*)d_in[12];
  const float* bo = (const float*)d_in[13];
  float* out = (float*)d_out;

  char* ws = (char*)d_ws;
  __bf16* Wt   = (__bf16*)(ws + OFF_WT);
  __bf16* Kbuf = (__bf16*)(ws + OFF_K);
  __bf16* Vbuf = (__bf16*)(ws + OFF_V);
  int*    qm   = (int*)(ws + OFF_QM);
  int*    km   = (int*)(ws + OFF_KM);
  int*    vm   = (int*)(ws + OFF_VM);
  float*  S    = (float*)(ws + OFF_S);
  float*  Z    = (float*)(ws + OFF_Z);

  prep_kernel<<<(3 * KSTEPS * U_ * 32 + 255) / 256, 256, 0, stream>>>(Wq, Wk, Wv, Wt, qm, km, vm);

  dim3 ggrid(L_ / MT, U_ / NT, B_);
  qkv_gemm<<<ggrid, 256, 0, stream>>>(t, Wt, bq, bk, bv, Kbuf, Vbuf, qm, km, vm);

  dim3 sgrid(L_ / 256, H_, B_);
  scores_kernel<<<sgrid, 256, 0, stream>>>(Kbuf, qm, t_mask, S);

  dim3 agrid(H_, B_);
  attn_kernel<<<agrid, 256, 0, stream>>>(Vbuf, qm, km, vm, g_mask, S, Z);

  final_kernel<<<B_, 256, 0, stream>>>(Z, Wh, bh, Wo, bo, out);
}
